// Attention_29326036697657
// MI455X (gfx1250) — matrix operand order repaired and checked by reading, not compiled, not run on hardware
//
#include <hip/hip_runtime.h>
#include <cstddef>

// ---------------------------------------------------------------------------
// CDNA5 (gfx1250) wave32 WMMA implementation of the dual-attention block.
// All heavy GEMMs use v_wmma_f32_16x16x32_bf16 with f32 accumulation.
// Fragments are loaded as contiguous b128 vectors (weights pre-transposed to
// [N][K]; attention V slot stored [b][h][d][n]) with k = 16*(lane/16) + s in both A and B.
// ---------------------------------------------------------------------------

typedef __bf16 bf16;
typedef __attribute__((ext_vector_type(16))) __bf16          v16bf;
typedef __attribute__((ext_vector_type(8)))  float           v8f;
typedef __attribute__((ext_vector_type(8)))  unsigned short  us8;
typedef __attribute__((ext_vector_type(4)))  unsigned short  us4;
typedef __attribute__((ext_vector_type(4)))  float           f4;

#define DIMC   768
#define HEADS  12
#define HD     64
#define NSEQ   144
#define BATCH  64
#define NQKV   3072
#define MROWS  (BATCH * NSEQ)   // 9216

static __device__ __forceinline__ unsigned short f2us(float f) {
    unsigned int u = __builtin_bit_cast(unsigned int, f);
    unsigned int r = (u + 0x7FFFu + ((u >> 16) & 1u)) >> 16;   // RNE
    return (unsigned short)r;
}
static __device__ __forceinline__ float us2f(unsigned short u) {
    unsigned int v = ((unsigned int)u) << 16;
    return __builtin_bit_cast(float, v);
}
static __device__ __forceinline__ v8f vzero8() {
    v8f z;
#pragma unroll
    for (int i = 0; i < 8; ++i) z[i] = 0.f;
    return z;
}
static __device__ __forceinline__ us8 uzero8() {
    us8 z;
#pragma unroll
    for (int i = 0; i < 8; ++i) z[i] = 0;
    return z;
}
static __device__ __forceinline__ v16bf mk16(us8 lo, us8 hi) {
    union { us8 h[2]; v16bf v; } u;
    u.h[0] = lo; u.h[1] = hi;
    return u.v;
}
// A fragment (16x32 MxK, 16-bit): this lane holds K {kc+khalf+0..15}.
// p = &src[m*ld + kc + khalf]  (16B aligned)  -> two b128 loads.
static __device__ __forceinline__ v16bf load_a_frag(const unsigned short* p) {
    return mk16(*(const us8*)p, *(const us8*)(p + 8));
}
// B fragment (32x16 KxN, 16-bit): this lane holds K {kc+ksec+0..15} of one col.
// p = &srcT[col*ld + kc + ksec]  (K-major source) -> two b128 loads.
static __device__ __forceinline__ v16bf load_b_frag(const unsigned short* p) {
    return mk16(*(const us8*)p, *(const us8*)(p + 8));
}

// ---------------------------------------------------------------------------
// f32 -> bf16 conversion, 4 elements/thread (b128 in, b64 out)
// ---------------------------------------------------------------------------
__global__ void cvt_f32_bf16_v4(const float* __restrict__ s,
                                unsigned short* __restrict__ d, int n4) {
    for (int i = blockIdx.x * blockDim.x + threadIdx.x; i < n4;
         i += gridDim.x * blockDim.x) {
        f4 v = *(const f4*)(s + 4 * (size_t)i);
        us4 o;
#pragma unroll
        for (int j = 0; j < 4; ++j) o[j] = f2us(v[j]);
        *(us4*)(d + 4 * (size_t)i) = o;
    }
}

// ---------------------------------------------------------------------------
// f32 [K x N] -> bf16 [N x K] transpose-convert (weights, one-time, tiny)
// ---------------------------------------------------------------------------
__global__ __launch_bounds__(256) void cvt_transpose(
    const float* __restrict__ src, unsigned short* __restrict__ dst,
    int K, int N) {
    __shared__ float tile[16][17];
    const int k0 = blockIdx.x * 16, n0 = blockIdx.y * 16;
    const int tx = threadIdx.x & 15, ty = threadIdx.x >> 4;
    tile[ty][tx] = src[(size_t)(k0 + ty) * N + n0 + tx];
    __syncthreads();
    dst[(size_t)(n0 + ty) * K + k0 + tx] = f2us(tile[tx][ty]);
}

// ---------------------------------------------------------------------------
// QKV GEMM: A[M x 768] bf16 @ Wt[3072 x 768] (pre-transposed) -> scatter into
//   q,k slots: [slot][b][h][n][64] ; v slot (which==3): [b][h][d][n] (for P@V).
// Block: 8 waves, 32 rows x 256 cols; wave = 32x32 tile (2x2 fragments).
// ---------------------------------------------------------------------------
__global__ __launch_bounds__(256) void gemm_qkv(
    const unsigned short* __restrict__ A, const unsigned short* __restrict__ Wt,
    unsigned short* __restrict__ QKV, int nslots) {
    __shared__ alignas(16) unsigned short As[32 * DIMC];   // 48 KB
    const int rowTile = blockIdx.x;                        // 0..287
    const int tid = threadIdx.x;
    const unsigned short* Arow = A + (size_t)rowTile * 32 * DIMC;
    for (int i = tid * 8; i < 32 * DIMC; i += 256 * 8)
        *(us8*)(As + i) = *(const us8*)(Arow + i);
    __syncthreads();

    const int wave = tid >> 5, lane = tid & 31;
    const int col0 = blockIdx.y * 256 + wave * 32;         // two 16-col tiles
    const int m = lane & 15, colN = lane & 15;
    const int khalf = (lane >> 4) << 4, ksec = (lane >> 4) << 4;

    v8f c[2][2];
#pragma unroll
    for (int rt = 0; rt < 2; ++rt)
#pragma unroll
        for (int bt = 0; bt < 2; ++bt) c[rt][bt] = vzero8();

    for (int kc = 0; kc < DIMC; kc += 32) {
        __builtin_prefetch(Wt + (size_t)(col0 + colN) * DIMC + kc + 64, 0, 0);
        v16bf a0 = load_a_frag(As + (0 * 16 + m) * DIMC + kc + khalf);
        v16bf a1 = load_a_frag(As + (1 * 16 + m) * DIMC + kc + khalf);
        v16bf b0 = load_b_frag(Wt + (size_t)(col0 + colN) * DIMC + kc + ksec);
        v16bf b1 = load_b_frag(Wt + (size_t)(col0 + 16 + colN) * DIMC + kc + ksec);
        c[0][0] = __builtin_amdgcn_wmma_f32_16x16x32_bf16(false, a0, false, b0, (short)0, c[0][0], false, false);
        c[0][1] = __builtin_amdgcn_wmma_f32_16x16x32_bf16(false, a0, false, b1, (short)0, c[0][1], false, false);
        c[1][0] = __builtin_amdgcn_wmma_f32_16x16x32_bf16(false, a1, false, b0, (short)0, c[1][0], false, false);
        c[1][1] = __builtin_amdgcn_wmma_f32_16x16x32_bf16(false, a1, false, b1, (short)0, c[1][1], false, false);
    }

#pragma unroll
    for (int bt = 0; bt < 2; ++bt) {
        const int cg = col0 + bt * 16 + colN;              // global column
        const int which = cg / DIMC;                       // 0..3
        if (which < 1 || which > nslots) continue;
        const int rem = cg % DIMC, hh = rem >> 6, dd = rem & 63;
#pragma unroll
        for (int rt = 0; rt < 2; ++rt)
#pragma unroll
            for (int r = 0; r < 8; ++r) {
                int row = rowTile * 32 + rt * 16 + r + ((lane >> 4) << 3);
                int bb = row / NSEQ, nn = row % NSEQ;
                size_t idx;
                if (which == 3)    // V transposed: [b][h][d][n]
                    idx = (size_t)2 * BATCH * HEADS * NSEQ * HD +
                          (((size_t)bb * HEADS + hh) * HD + dd) * NSEQ + nn;
                else               // Q/K: [slot][b][h][n][d]
                    idx = (((size_t)(which - 1) * BATCH + bb) * HEADS + hh) * (size_t)NSEQ * HD +
                          (size_t)nn * HD + dd;
                QKV[idx] = f2us(c[rt][bt][r]);
            }
    }
}

// ---------------------------------------------------------------------------
// Bias GEMM: A[M x 768] bf16 @ Wt[768 x 768] (pre-transposed) + bias
//   -> f32 out (+ optional bf16 mirror). Same 32x256 block / 32x32 wave tiling.
// ---------------------------------------------------------------------------
__global__ __launch_bounds__(256) void gemm_proj(
    const unsigned short* __restrict__ A, const unsigned short* __restrict__ Wt,
    const float* __restrict__ bias, float* __restrict__ Cf,
    unsigned short* __restrict__ Cb) {
    __shared__ alignas(16) unsigned short As[32 * DIMC];
    const int rowTile = blockIdx.x;
    const int tid = threadIdx.x;
    const unsigned short* Arow = A + (size_t)rowTile * 32 * DIMC;
    for (int i = tid * 8; i < 32 * DIMC; i += 256 * 8)
        *(us8*)(As + i) = *(const us8*)(Arow + i);
    __syncthreads();

    const int wave = tid >> 5, lane = tid & 31;
    const int col0 = blockIdx.y * 256 + wave * 32;
    const int m = lane & 15, colN = lane & 15;
    const int khalf = (lane >> 4) << 4, ksec = (lane >> 4) << 4;

    v8f c[2][2];
#pragma unroll
    for (int rt = 0; rt < 2; ++rt)
#pragma unroll
        for (int bt = 0; bt < 2; ++bt) c[rt][bt] = vzero8();

    for (int kc = 0; kc < DIMC; kc += 32) {
        __builtin_prefetch(Wt + (size_t)(col0 + colN) * DIMC + kc + 64, 0, 0);
        v16bf a0 = load_a_frag(As + (0 * 16 + m) * DIMC + kc + khalf);
        v16bf a1 = load_a_frag(As + (1 * 16 + m) * DIMC + kc + khalf);
        v16bf b0 = load_b_frag(Wt + (size_t)(col0 + colN) * DIMC + kc + ksec);
        v16bf b1 = load_b_frag(Wt + (size_t)(col0 + 16 + colN) * DIMC + kc + ksec);
        c[0][0] = __builtin_amdgcn_wmma_f32_16x16x32_bf16(false, a0, false, b0, (short)0, c[0][0], false, false);
        c[0][1] = __builtin_amdgcn_wmma_f32_16x16x32_bf16(false, a0, false, b1, (short)0, c[0][1], false, false);
        c[1][0] = __builtin_amdgcn_wmma_f32_16x16x32_bf16(false, a1, false, b0, (short)0, c[1][0], false, false);
        c[1][1] = __builtin_amdgcn_wmma_f32_16x16x32_bf16(false, a1, false, b1, (short)0, c[1][1], false, false);
    }

#pragma unroll
    for (int bt = 0; bt < 2; ++bt) {
        const int col = col0 + bt * 16 + colN;
        const float bv = bias[col];
#pragma unroll
        for (int rt = 0; rt < 2; ++rt)
#pragma unroll
            for (int r = 0; r < 8; ++r) {
                int row = rowTile * 32 + rt * 16 + r + ((lane >> 4) << 3);
                float val = c[rt][bt][r] + bv;
                Cf[(size_t)row * DIMC + col] = val;
                if (Cb) Cb[(size_t)row * DIMC + col] = f2us(val);
            }
    }
}

// ---------------------------------------------------------------------------
// SE layer folded in-place:  q *= (1 + sig(relu(mean@w1)@w2) + sig(relu(max@w1)@w2))
// One block per batch element; q layout [b][h][n][64].
// ---------------------------------------------------------------------------
__global__ __launch_bounds__(256) void se_kernel(unsigned short* __restrict__ q,
                                                 const float* __restrict__ w1,
                                                 const float* __restrict__ w2) {
    __shared__ float pmean[DIMC], pmax[DIMC], hid[2][48], gate[DIMC];
    const int b = blockIdx.x, tid = threadIdx.x;
    unsigned short* qb = q + (size_t)b * HEADS * NSEQ * HD;

    for (int c = tid; c < DIMC; c += 256) {
        int h = c >> 6, d = c & 63;
        float s = 0.f, mx = -3.0e38f;
        for (int n = 0; n < NSEQ; ++n) {
            float v = us2f(qb[((size_t)h * NSEQ + n) * HD + d]);
            s += v; mx = fmaxf(mx, v);
        }
        pmean[c] = s * (1.f / NSEQ); pmax[c] = mx;
    }
    __syncthreads();
    if (tid < 96) {
        int j = tid % 48, sel = tid / 48;
        const float* src = sel ? pmax : pmean;
        float acc = 0.f;
        for (int c = 0; c < DIMC; ++c) acc += src[c] * w1[c * 48 + j];
        hid[sel][j] = fmaxf(acc, 0.f);
    }
    __syncthreads();
    for (int c = tid; c < DIMC; c += 256) {
        float a0 = 0.f, a1 = 0.f;
        for (int j = 0; j < 48; ++j) {
            float w = w2[j * DIMC + c];
            a0 += hid[0][j] * w; a1 += hid[1][j] * w;
        }
        gate[c] = 1.f + 1.f / (1.f + __expf(-a0)) + 1.f / (1.f + __expf(-a1));
    }
    __syncthreads();
    for (int i = tid; i < HEADS * NSEQ * HD; i += 256) {
        int c = (i / (NSEQ * HD)) * 64 + (i & 63);
        qb[i] = f2us(us2f(qb[i]) * gate[c]);
    }
}

// ---------------------------------------------------------------------------
// SA layer folded in-place: q *= (1 + sigmoid(conv5x5([mean_c;max_c]) + b)) per n.
// ---------------------------------------------------------------------------
__global__ __launch_bounds__(256) void sa_kernel(unsigned short* __restrict__ q,
                                                 const float* __restrict__ w,
                                                 const float* __restrict__ bias) {
    __shared__ float cmean[NSEQ], cmax[NSEQ], gate[NSEQ];
    const int b = blockIdx.x, tid = threadIdx.x;
    unsigned short* qb = q + (size_t)b * HEADS * NSEQ * HD;

    if (tid < NSEQ) {
        float s = 0.f, mx = -3.0e38f;
        for (int c = 0; c < DIMC; ++c) {
            int h = c >> 6, d = c & 63;
            float v = us2f(qb[((size_t)h * NSEQ + tid) * HD + d]);
            s += v; mx = fmaxf(mx, v);
        }
        cmean[tid] = s * (1.f / DIMC); cmax[tid] = mx;
    }
    __syncthreads();
    if (tid < NSEQ) {
        int py = tid / 12, px = tid % 12;
        float acc = bias[0];
        for (int ky = 0; ky < 5; ++ky)
            for (int kx = 0; kx < 5; ++kx) {
                int iy = py + ky - 2, ix = px + kx - 2;
                if (iy >= 0 && iy < 12 && ix >= 0 && ix < 12) {
                    int p = iy * 12 + ix;
                    acc += cmean[p] * w[ky * 5 + kx] + cmax[p] * w[25 + ky * 5 + kx];
                }
            }
        gate[tid] = 1.f + 1.f / (1.f + __expf(-acc));
    }
    __syncthreads();
    for (int i = tid; i < HEADS * NSEQ * HD; i += 256) {
        int n = (i >> 6) % NSEQ;
        qb[i] = f2us(us2f(qb[i]) * gate[n]);
    }
}

// ---------------------------------------------------------------------------
// Fused attention: one wave per (b, h, 16-row strip).
// S = (Q K^T)*scale -> softmax (16-lane shuffle reductions) -> P bf16 in LDS
// -> O = P @ V with V pre-transposed [d][n] so B fragments are contiguous.
// Q already carries (1+gate). Output written [b][n][c] bf16 for proj GEMMs.
// ---------------------------------------------------------------------------
__global__ __launch_bounds__(32) void attn_kernel(
    const unsigned short* __restrict__ Q, const unsigned short* __restrict__ K,
    const unsigned short* __restrict__ VT, unsigned short* __restrict__ O) {
    __shared__ alignas(16) unsigned short P[16 * 160];     // K padded 144 -> 160
    const int blk = blockIdx.x;                            // b*108 + h*9 + s
    const int s = blk % 9, hh = (blk / 9) % HEADS, bb = blk / (9 * HEADS);
    const int lane = threadIdx.x;
    const unsigned short* q  = Q  + ((size_t)bb * HEADS + hh) * NSEQ * HD;
    const unsigned short* k  = K  + ((size_t)bb * HEADS + hh) * NSEQ * HD;
    const unsigned short* vt = VT + ((size_t)bb * HEADS + hh) * HD * NSEQ; // [d][n]
    const int m0 = s * 16;
    const int m = lane & 15, colN = lane & 15;
    const int khalf = (lane >> 4) << 4, ksec = (lane >> 4) << 4;

    for (int i = lane * 8; i < 16 * 160; i += 32 * 8)
        *(us8*)(P + i) = uzero8();                         // zero (incl. pad cols)

    v8f cs[9];
#pragma unroll
    for (int t = 0; t < 9; ++t) cs[t] = vzero8();

#pragma unroll
    for (int kc = 0; kc < HD; kc += 32) {
        v16bf a = load_a_frag(q + (size_t)(m0 + m) * HD + kc + khalf);
#pragma unroll
        for (int t = 0; t < 9; ++t) {
            v16bf bfr = load_b_frag(k + (size_t)(t * 16 + colN) * HD + kc + ksec);
            cs[t] = __builtin_amdgcn_wmma_f32_16x16x32_bf16(false, a, false, bfr,
                                                            (short)0, cs[t], false, false);
        }
    }

    const float scale = 0.125f;                            // hd^-0.5
#pragma unroll
    for (int r = 0; r < 8; ++r) {                          // row m = r + 8*(lane>=16)
        float mx = -3.0e38f;
#pragma unroll
        for (int t = 0; t < 9; ++t) { cs[t][r] *= scale; mx = fmaxf(mx, cs[t][r]); }
#pragma unroll
        for (int off = 1; off < 16; off <<= 1) mx = fmaxf(mx, __shfl_xor(mx, off, 32));
        float sum = 0.f;
#pragma unroll
        for (int t = 0; t < 9; ++t) { float e = __expf(cs[t][r] - mx); cs[t][r] = e; sum += e; }
#pragma unroll
        for (int off = 1; off < 16; off <<= 1) sum += __shfl_xor(sum, off, 32);
        float inv = 1.f / sum;
        int row = r + ((lane >> 4) << 3);
#pragma unroll
        for (int t = 0; t < 9; ++t) P[row * 160 + t * 16 + colN] = f2us(cs[t][r] * inv);
    }
    __syncthreads();

    // O strip = P(16x160) @ V(160x64); padded P columns are zero, so the
    // 16-element overreads of VT rows contribute nothing.
#pragma unroll
    for (int tc = 0; tc < 4; ++tc) {
        v8f co = vzero8();
#pragma unroll
        for (int kc = 0; kc < 160; kc += 32) {
            v16bf a   = load_a_frag(P + m * 160 + kc + khalf);
            v16bf bfr = load_b_frag(vt + (size_t)(tc * 16 + colN) * NSEQ + kc + ksec);
            co = __builtin_amdgcn_wmma_f32_16x16x32_bf16(false, a, false, bfr,
                                                         (short)0, co, false, false);
        }
#pragma unroll
        for (int r = 0; r < 8; ++r) {
            int row = m0 + r + ((lane >> 4) << 3);
            int cidx = hh * HD + tc * 16 + colN;
            O[((size_t)bb * NSEQ + row) * DIMC + cidx] = f2us(co[r]);
        }
    }
}

// ---------------------------------------------------------------------------
// Host launch
// ---------------------------------------------------------------------------
extern "C" void kernel_launch(void* const* d_in, const int* in_sizes, int n_in,
                              void* d_out, int out_size, void* d_ws, size_t ws_size,
                              hipStream_t stream) {
    const float* x     = (const float*)d_in[0];
    const float* y     = (const float*)d_in[1];
    const float* Wqkv  = (const float*)d_in[2];
    const float* Wproj = (const float*)d_in[3];
    const float* bproj = (const float*)d_in[4];
    const float* se_w1 = (const float*)d_in[5];
    const float* se_w2 = (const float*)d_in[6];
    const float* sa_w  = (const float*)d_in[7];
    const float* sa_b  = (const float*)d_in[8];
    float* out = (float*)d_out;

    constexpr size_t E_X    = (size_t)MROWS * DIMC;     // 7,077,888
    constexpr size_t E_WQKV = (size_t)DIMC * NQKV;
    constexpr size_t E_WPRJ = (size_t)DIMC * DIMC;
    constexpr size_t O_XB   = 0;
    constexpr size_t O_YB   = O_XB + E_X * 2;
    constexpr size_t O_WQ   = O_YB + E_X * 2;           // Wqkv^T  [3072 x 768]
    constexpr size_t O_WP   = O_WQ + E_WQKV * 2;        // Wproj^T [768 x 768]
    constexpr size_t O_QKVX = O_WP + E_WPRJ * 2;        // q,k [b,h,n,d]; v [b,h,d,n]
    constexpr size_t O_QKVY = O_QKVX + 3 * E_X * 2 + 256;  // pad for VT overread
    constexpr size_t O_OUT1 = O_QKVY + 2 * E_X * 2;
    constexpr size_t O_OUT2 = O_OUT1 + E_X * 2;
    constexpr size_t O_X1B  = O_OUT2 + E_X * 2;
    constexpr size_t O_Y1B  = O_X1B + E_X * 2;

    char* ws = (char*)d_ws;
    unsigned short* xb   = (unsigned short*)(ws + O_XB);
    unsigned short* yb   = (unsigned short*)(ws + O_YB);
    unsigned short* wqT  = (unsigned short*)(ws + O_WQ);
    unsigned short* wpT  = (unsigned short*)(ws + O_WP);
    unsigned short* qkvx = (unsigned short*)(ws + O_QKVX);
    unsigned short* qkvy = (unsigned short*)(ws + O_QKVY);
    unsigned short* out1 = (unsigned short*)(ws + O_OUT1);
    unsigned short* out2 = (unsigned short*)(ws + O_OUT2);
    unsigned short* x1b  = (unsigned short*)(ws + O_X1B);
    unsigned short* y1b  = (unsigned short*)(ws + O_Y1B);

    unsigned short* qx  = qkvx;                 // slot 0: q  [b,h,n,d]
    unsigned short* kx  = qkvx + E_X;           // slot 1: k  [b,h,n,d]
    unsigned short* vxT = qkvx + 2 * E_X;       // slot 2: v^T [b,h,d,n]
    unsigned short* qy  = qkvy;
    unsigned short* ky  = qkvy + E_X;

    // 1) precision conversion (+ weight transposes for K-contiguous B frags)
    cvt_f32_bf16_v4<<<1024, 256, 0, stream>>>(x, xb, (int)(E_X / 4));
    cvt_f32_bf16_v4<<<1024, 256, 0, stream>>>(y, yb, (int)(E_X / 4));
    cvt_transpose<<<dim3(DIMC / 16, NQKV / 16), 256, 0, stream>>>(Wqkv, wqT, DIMC, NQKV);
    cvt_transpose<<<dim3(DIMC / 16, DIMC / 16), 256, 0, stream>>>(Wproj, wpT, DIMC, DIMC);

    // 2) QKV projections (which==0 columns dead; y needs only q,k)
    gemm_qkv<<<dim3(MROWS / 32, NQKV / 256), 256, 0, stream>>>(xb, wqT, qkvx, 3);
    gemm_qkv<<<dim3(MROWS / 32, NQKV / 256), 256, 0, stream>>>(yb, wqT, qkvy, 2);

    // 3) gates folded in place:  q <- q * (1 + gate)   (dots fusion)
    se_kernel<<<BATCH, 256, 0, stream>>>(qx, se_w1, se_w2);
    sa_kernel<<<BATCH, 256, 0, stream>>>(qy, sa_w, sa_b);

    // 4) attentions (vy == vx per reference bug; attnx/attny are dead code)
    attn_kernel<<<BATCH * HEADS * 9, 32, 0, stream>>>(qx, kx, vxT, out1);
    attn_kernel<<<BATCH * HEADS * 9, 32, 0, stream>>>(qy, ky, vxT, out2);

    // 5) projections: x1, y1 (bf16 mirrors), then xo=proj(x1), yo=proj(y1)
    const size_t S = E_X;
    gemm_proj<<<dim3(MROWS / 32, DIMC / 256), 256, 0, stream>>>(out1, wpT, bproj, out + 0 * S, x1b);
    gemm_proj<<<dim3(MROWS / 32, DIMC / 256), 256, 0, stream>>>(out2, wpT, bproj, out + 1 * S, y1b);
    gemm_proj<<<dim3(MROWS / 32, DIMC / 256), 256, 0, stream>>>(x1b,  wpT, bproj, out + 2 * S, nullptr);
    gemm_proj<<<dim3(MROWS / 32, DIMC / 256), 256, 0, stream>>>(y1b,  wpT, bproj, out + 3 * S, nullptr);
}
